// FLASH_ShareA_FFConvM_44032004718634
// MI455X (gfx1250) — matrix operand order repaired and checked by reading, not compiled, not run on hardware
//
#include <hip/hip_runtime.h>
#include <hip/hip_bf16.h>

// ---------------- problem constants (match reference) ----------------
#define DIM    512
#define HID    2048
#define QKD    128
#define GRP    256
#define KW     17
#define SEQ    4096
#define BATCH  4
#define TOK    (BATCH * SEQ)      // 16384 tokens
#define GN     (SEQ / GRP)        // 16 groups per batch
#define HHALF  (HID / 2)          // 1024 = 2*DIM

// ---------------- CDNA5 WMMA types ----------------
typedef __attribute__((ext_vector_type(16))) __bf16 v16bf;
typedef __attribute__((ext_vector_type(8)))  float  v8f;

union Frag16 {
    uint4          q[2];   // 32 bytes = two b128 LDS loads
    unsigned short us[16];
    v16bf          v;
};

__device__ __forceinline__ unsigned short f2bf(float f) {
    union { float f; unsigned u; } a; a.f = f;
    unsigned u = a.u + 0x7FFFu + ((a.u >> 16) & 1u);   // round-to-nearest-even
    return (unsigned short)(u >> 16);
}
__device__ __forceinline__ float sigm(float x) { return 1.f / (1.f + __expf(-x)); }

// CDNA5 async global->LDS copy (ASYNCcnt-tracked, bypasses VGPRs).
// lds_off = wave-relative LDS byte offset (low 32 bits of a __shared__ pointer).
__device__ __forceinline__ void async_g2l_b128(unsigned lds_off, const void* gaddr) {
    asm volatile("global_load_async_to_lds_b128 %0, %1, off"
                 :: "v"(lds_off), "v"(gaddr) : "memory");
}
__device__ __forceinline__ void wait_async0() {
    asm volatile("s_wait_asynccnt 0x0" ::: "memory");
}

// ---------------- generic WMMA GEMM ----------------
// C[M,N] = epilogue( A[M,K] @ B[K,N] )
//   TA: A[m][k] = Asrc[k*lda + m]  (else Asrc[m*lda + k])
//   TB: B[k][n] = Bsrc[n*ldb + k]  (else Bsrc[k*ldb + n])
// blockIdx.z batches via element strides strA/strB/strC.
enum { EPI_BIAS_SILU = 0, EPI_F32 = 1, EPI_ACC_F32 = 2, EPI_RELU2_BF16 = 3, EPI_BF16 = 4 };

template <bool TA, bool TB, int EPI>
__global__ void __launch_bounds__(256)
k_gemm(const unsigned short* __restrict__ A, const unsigned short* __restrict__ B,
       const float* __restrict__ bias, float* __restrict__ Cf, unsigned short* __restrict__ Cb,
       int M, int N, int K, int lda, int ldb, int ldc, float alpha,
       long long strA, long long strB, long long strC)
{
    constexpr int BM = 128, BN = 64, BK = 32;
    constexpr int SA = BK + 8;   // padded LDS strides keep 16B alignment (80B / 144B rows)
    constexpr int SB = BN + 8;
    __shared__ unsigned short sA[BM * SA];
    __shared__ unsigned short sB[BK * SB];

    const int tid  = threadIdx.x;
    const int lane = tid & 31;
    const int wave = tid >> 5;                 // 8 waves: wave w owns rows [16w,16w+16)
    const int m0   = blockIdx.y * BM;
    const int n0   = blockIdx.x * BN;
    const long long z = blockIdx.z;
    A += z * strA;
    B += z * strB;

    v8f acc[4] = {};

    for (int k0 = 0; k0 < K; k0 += BK) {
        // ---- stage A tile: BM x BK (256 thr x 16 elems) ----
        {
            const int r = tid >> 1, seg = tid & 1;
            if (!TA) {
                const unsigned short* ap = A + (size_t)(m0 + r) * lda + (k0 + seg * 16);
                async_g2l_b128((unsigned)(size_t)&sA[r * SA + seg * 16    ], ap);
                async_g2l_b128((unsigned)(size_t)&sA[r * SA + seg * 16 + 8], ap + 8);
            } else {
                #pragma unroll
                for (int j = 0; j < 16; ++j) {
                    const int c = seg * 16 + j;
                    sA[r * SA + c] = A[(size_t)(k0 + c) * lda + (m0 + r)];
                }
            }
        }
        // ---- stage B tile: BK x BN (256 thr x 8 elems) ----
        {
            const int r = tid >> 3, seg = tid & 7;
            if (!TB) {
                const unsigned short* bp = B + (size_t)(k0 + r) * ldb + (n0 + seg * 8);
                async_g2l_b128((unsigned)(size_t)&sB[r * SB + seg * 8], bp);
            } else {
                #pragma unroll
                for (int j = 0; j < 8; ++j) {
                    const int n = seg * 8 + j;
                    sB[r * SB + n] = B[(size_t)(n0 + n) * ldb + (k0 + r)];
                }
            }
        }
        if (!TA || !TB) wait_async0();   // async LDS fills complete before the barrier
        __syncthreads();

        // ---- A fragment (16-bit 16x32 layout: lane<16 -> K {0..7,16..23}; lane>=16 -> +8) ----
        Frag16 fa;
        const int mrow = wave * 16 + (lane & 15);
        const int kb   = (lane >> 4) * 8;
        fa.q[0] = *reinterpret_cast<const uint4*>(&sA[mrow * SA + kb]);
        fa.q[1] = *reinterpret_cast<const uint4*>(&sA[mrow * SA + kb + 16]);

        #pragma unroll
        for (int c = 0; c < 4; ++c) {
            // B fragment: lane%16 = N column, K {0..7,16..23} (+8 for lane>=16)
            Frag16 fb;
            #pragma unroll
            for (int s = 0; s < 16; ++s)
                fb.us[s] = sB[(kb + (s & 7) + ((s >> 3) << 4)) * SB + c * 16 + (lane & 15)];
            acc[c] = __builtin_amdgcn_wmma_f32_16x16x32_bf16(
                false, fa.v, false, fb.v, (short)0, acc[c], false, false);
        }
        __syncthreads();
    }

    // ---- epilogue: D layout, VGPR r -> M = r + 8*(lane>=16), N = lane&15 ----
    #pragma unroll
    for (int c = 0; c < 4; ++c) {
        #pragma unroll
        for (int r = 0; r < 8; ++r) {
            const int gm = m0 + wave * 16 + r + ((lane >> 4) << 3);
            const int gn = n0 + c * 16 + (lane & 15);
            const size_t idx = (size_t)z * strC + (size_t)gm * ldc + gn;
            const float vv = acc[c][r];
            if (EPI == EPI_BIAS_SILU)      { const float t = vv + bias[gn]; Cf[idx] = t * sigm(t); }
            else if (EPI == EPI_F32)         Cf[idx] = vv * alpha;
            else if (EPI == EPI_ACC_F32)     Cf[idx] += vv * alpha;
            else if (EPI == EPI_RELU2_BF16){ float t = vv * alpha; t = t > 0.f ? t : 0.f; Cb[idx] = f2bf(t * t); }
            else                             Cb[idx] = f2bf(vv * alpha);
        }
    }
}

// ---------------- elementwise kernels ----------------

// token shift: first half of channels shifted right by one timestep (per batch)
__global__ void k_shift(const float* __restrict__ x, float* __restrict__ nx)
{
    const long long idx = (long long)blockIdx.x * 256 + threadIdx.x;   // over TOK*DIM
    const int c  = (int)(idx % DIM);
    const long long tg = idx / DIM;
    const int tl = (int)(tg % SEQ);
    float v;
    if (c < DIM / 2) v = (tl > 0) ? x[idx - DIM] : 0.f;
    else             v = x[idx];
    nx[idx] = v;
}

// LayerNorm over last dim C, output bf16 (WMMA A operand)
__global__ void k_layernorm_bf16(const float* __restrict__ x, const float* __restrict__ g,
                                 const float* __restrict__ bv, unsigned short* __restrict__ out, int C)
{
    __shared__ float red[256];
    const int tid = threadIdx.x;
    const float* xr = x + (size_t)blockIdx.x * C;
    float s = 0.f;
    for (int c = tid; c < C; c += 256) s += xr[c];
    red[tid] = s; __syncthreads();
    for (int o = 128; o > 0; o >>= 1) { if (tid < o) red[tid] += red[tid + o]; __syncthreads(); }
    const float mean = red[0] * (1.f / C);
    __syncthreads();
    float vs = 0.f;
    for (int c = tid; c < C; c += 256) { const float d = xr[c] - mean; vs += d * d; }
    red[tid] = vs; __syncthreads();
    for (int o = 128; o > 0; o >>= 1) { if (tid < o) red[tid] += red[tid + o]; __syncthreads(); }
    const float rstd = rsqrtf(red[0] * (1.f / C) + 1e-5f);
    unsigned short* orow = out + (size_t)blockIdx.x * C;
    for (int c = tid; c < C; c += 256)
        orow[c] = f2bf((xr[c] - mean) * rstd * g[c] + bv[c]);
}

__global__ void k_f2bf(const float* __restrict__ src, unsigned short* __restrict__ dst, long long n)
{
    const long long i = (long long)blockIdx.x * 256 + threadIdx.x;
    if (i < n) dst[i] = f2bf(src[i]);
}

// naive 17-tap depthwise conv (used only for the small qk path, C=128)
__global__ void k_dwconv(const float* __restrict__ h, const float* __restrict__ kern,
                         const float* __restrict__ resid, float* __restrict__ out,
                         int C, long long total)
{
    const long long idx = (long long)blockIdx.x * 256 + threadIdx.x;
    if (idx >= total) return;
    const int c = (int)(idx % C);
    const long long tg = idx / C;
    const int tl = (int)(tg % SEQ);
    const long long bb = tg / SEQ;
    float acc = h[idx];
    float s = 0.f;
    #pragma unroll
    for (int kw = 0; kw < KW; ++kw) {
        const int tt = tl + kw - KW / 2;
        if (tt >= 0 && tt < SEQ)
            s += h[(bb * SEQ + tt) * (long long)C + c] * kern[kw * C + c];
    }
    acc += s;
    if (resid) acc += resid[idx];
    out[idx] = acc;
}

// LDS-tiled 17-tap depthwise conv: TT timesteps + 16 halo x 256 channels per block.
// out = h + dwconv(h) (+resid). Optionally fuses the bf16 v/u split (hid path).
// grid: (C/256, SEQ/TT, BATCH); C must be a multiple of 256.
template <int TT>
__global__ void __launch_bounds__(256)
k_dwconv_tiled(const float* __restrict__ h, const float* __restrict__ kern,
               const float* __restrict__ resid, float* __restrict__ out,
               unsigned short* __restrict__ vb, unsigned short* __restrict__ ub, int C)
{
    __shared__ float tile[(TT + 16) * 256];
    const int tid = threadIdx.x;
    const int c   = blockIdx.x * 256 + tid;
    const int t0  = blockIdx.y * TT;
    const long long bb = blockIdx.z;
    const long long base = bb * SEQ * (long long)C;

    #pragma unroll 4
    for (int r = 0; r < TT + 16; ++r) {
        const int tt = t0 - 8 + r;
        tile[r * 256 + tid] = (tt >= 0 && tt < SEQ) ? h[base + (long long)tt * C + c] : 0.f;
    }
    __syncthreads();

    float kc[KW];
    #pragma unroll
    for (int kw = 0; kw < KW; ++kw) kc[kw] = kern[kw * C + c];

    #pragma unroll
    for (int t = 0; t < TT; ++t) {
        float s = tile[(t + 8) * 256 + tid];       // identity term
        float a = 0.f;
        #pragma unroll
        for (int kw = 0; kw < KW; ++kw) a += tile[(t + kw) * 256 + tid] * kc[kw];
        s += a;
        const long long idx = base + (long long)(t0 + t) * C + c;
        if (resid) s += resid[idx];
        out[idx] = s;
        if (vb) {  // fused v/u bf16 split (C == HID)
            const long long tok = bb * SEQ + t0 + t;
            if (c < HHALF) vb[tok * HHALF + c] = f2bf(s);
            else           ub[tok * HHALF + (c - HHALF)] = f2bf(s);
        }
    }
}

// offset-scale: 4 projections of qk, output bf16
__global__ void k_offscale(const float* __restrict__ qkf, const float* __restrict__ g,
                           const float* __restrict__ b,
                           unsigned short* __restrict__ qq, unsigned short* __restrict__ lq,
                           unsigned short* __restrict__ qk, unsigned short* __restrict__ lk)
{
    const long long idx = (long long)blockIdx.x * 256 + threadIdx.x;   // over TOK*QKD
    const int d = (int)(idx & (QKD - 1));
    const float v = qkf[idx];
    qq[idx] = f2bf(v * g[0 * QKD + d] + b[0 * QKD + d]);
    lq[idx] = f2bf(v * g[1 * QKD + d] + b[1 * QKD + d]);
    qk[idx] = f2bf(v * g[2 * QKD + d] + b[2 * QKD + d]);
    lk[idx] = f2bf(v * g[3 * QKD + d] + b[3 * QKD + d]);
}

// out = att_u * v * sigmoid(att_v * u)
__global__ void k_gate(const float* __restrict__ attv, const float* __restrict__ attu,
                       const float* __restrict__ h2, float* __restrict__ outg)
{
    const long long idx = (long long)blockIdx.x * 256 + threadIdx.x;   // over TOK*HHALF
    const long long tok = idx >> 10;
    const int c = (int)(idx & (HHALF - 1));
    const float v = h2[tok * HID + c];
    const float u = h2[tok * HID + HHALF + c];
    outg[idx] = attu[idx] * v * sigm(attv[idx] * u);
}

// ---------------- launch ----------------
extern "C" void kernel_launch(void* const* d_in, const int* in_sizes, int n_in,
                              void* d_out, int out_size, void* d_ws, size_t ws_size,
                              hipStream_t stream)
{
    (void)in_sizes; (void)n_in; (void)out_size; (void)ws_size;
    const float* x       = (const float*)d_in[0];
    const float* hid_lng = (const float*)d_in[1];
    const float* hid_lnb = (const float*)d_in[2];
    const float* hid_W   = (const float*)d_in[3];
    const float* hid_b   = (const float*)d_in[4];
    const float* hid_dwk = (const float*)d_in[5];
    const float* qk_lng  = (const float*)d_in[6];
    const float* qk_lnb  = (const float*)d_in[7];
    const float* qk_W    = (const float*)d_in[8];
    const float* qk_b    = (const float*)d_in[9];
    const float* qk_dwk  = (const float*)d_in[10];
    const float* os_g    = (const float*)d_in[11];
    const float* os_b    = (const float*)d_in[12];
    const float* out_lng = (const float*)d_in[13];
    const float* out_lnb = (const float*)d_in[14];
    const float* out_W   = (const float*)d_in[15];
    const float* out_b   = (const float*)d_in[16];
    const float* out_dwk = (const float*)d_in[17];
    float* y = (float*)d_out;

    char* p = (char*)d_ws;
    auto carve = [&](size_t bytes) -> void* {
        void* r = (void*)p;
        p += (bytes + 255) & ~(size_t)255;
        return r;
    };
    float*          nx   = (float*)carve((size_t)TOK * DIM * 4);
    unsigned short* lnh  = (unsigned short*)carve((size_t)TOK * DIM * 2);
    unsigned short* lnq  = (unsigned short*)carve((size_t)TOK * DIM * 2);
    unsigned short* whb  = (unsigned short*)carve((size_t)DIM * HID * 2);
    unsigned short* wqb  = (unsigned short*)carve((size_t)DIM * QKD * 2);
    unsigned short* wob  = (unsigned short*)carve((size_t)HHALF * DIM * 2);
    float*          hpre = (float*)carve((size_t)TOK * HID * 4);
    float*          h2   = (float*)carve((size_t)TOK * HID * 4);
    unsigned short* vb   = (unsigned short*)carve((size_t)TOK * HHALF * 2);
    unsigned short* ub   = (unsigned short*)carve((size_t)TOK * HHALF * 2);
    float*          qpre = (float*)carve((size_t)TOK * QKD * 4);
    float*          qkf  = (float*)carve((size_t)TOK * QKD * 4);
    unsigned short* pqq  = (unsigned short*)carve((size_t)TOK * QKD * 2);
    unsigned short* plq  = (unsigned short*)carve((size_t)TOK * QKD * 2);
    unsigned short* pqk  = (unsigned short*)carve((size_t)TOK * QKD * 2);
    unsigned short* plk  = (unsigned short*)carve((size_t)TOK * QKD * 2);
    unsigned short* attn = (unsigned short*)carve((size_t)BATCH * GN * GRP * GRP * 2);
    unsigned short* kvv  = (unsigned short*)carve((size_t)BATCH * QKD * HHALF * 2);
    unsigned short* kvu  = (unsigned short*)carve((size_t)BATCH * QKD * HHALF * 2);
    float*          attv = (float*)carve((size_t)TOK * HHALF * 4);
    float*          attu = (float*)carve((size_t)TOK * HHALF * 4);
    float*          outg = (float*)carve((size_t)TOK * HHALF * 4);
    unsigned short* lno  = (unsigned short*)carve((size_t)TOK * HHALF * 2);
    float*          g3   = (float*)carve((size_t)TOK * DIM * 4);

    const dim3 B256(256);

    // 1) token shift
    k_shift<<<dim3((TOK * DIM) / 256), B256, 0, stream>>>(x, nx);

    // 2) two LayerNorms of nx (bf16 outputs)
    k_layernorm_bf16<<<dim3(TOK), B256, 0, stream>>>(nx, hid_lng, hid_lnb, lnh, DIM);
    k_layernorm_bf16<<<dim3(TOK), B256, 0, stream>>>(nx, qk_lng, qk_lnb, lnq, DIM);

    // 3) weights -> bf16
    k_f2bf<<<dim3(((long long)DIM * HID + 255) / 256), B256, 0, stream>>>(hid_W, whb, (long long)DIM * HID);
    k_f2bf<<<dim3(((long long)DIM * QKD + 255) / 256), B256, 0, stream>>>(qk_W, wqb, (long long)DIM * QKD);
    k_f2bf<<<dim3(((long long)HHALF * DIM + 255) / 256), B256, 0, stream>>>(out_W, wob, (long long)HHALF * DIM);

    // 4) hid GEMM: silu(lnh @ hid_W + b)  (16384x2048x512)
    k_gemm<false, false, EPI_BIAS_SILU><<<dim3(HID / 64, TOK / 128, 1), B256, 0, stream>>>(
        lnh, whb, hid_b, hpre, nullptr, TOK, HID, DIM, DIM, HID, HID, 1.f, 0, 0, 0);

    // 5) hid ConvModule (LDS-tiled) + fused v/u bf16 split: h2 = hpre + dwconv(hpre)
    k_dwconv_tiled<16><<<dim3(HID / 256, SEQ / 16, BATCH), B256, 0, stream>>>(
        hpre, hid_dwk, nullptr, h2, vb, ub, HID);

    // 6) qk GEMM: silu(lnq @ qk_W + b)  (16384x128x512)
    k_gemm<false, false, EPI_BIAS_SILU><<<dim3(QKD / 64, TOK / 128, 1), B256, 0, stream>>>(
        lnq, wqb, qk_b, qpre, nullptr, TOK, QKD, DIM, DIM, QKD, QKD, 1.f, 0, 0, 0);

    // 7) qk ConvModule (small; naive)
    k_dwconv<<<dim3(((long long)TOK * QKD + 255) / 256), B256, 0, stream>>>(
        qpre, qk_dwk, nullptr, qkf, QKD, (long long)TOK * QKD);

    // 8) offset-scale -> 4 bf16 projections
    k_offscale<<<dim3((TOK * QKD) / 256), B256, 0, stream>>>(qkf, os_g, os_b, pqq, plq, pqk, plk);

    // 9) quad attention probs: attn = relu(qq @ qk^T / G)^2 -> bf16, per (batch,group)
    k_gemm<false, true, EPI_RELU2_BF16><<<dim3(GRP / 64, GRP / 128, BATCH * GN), B256, 0, stream>>>(
        pqq, pqk, nullptr, nullptr, attn, GRP, GRP, QKD, QKD, QKD, GRP, 1.f / GRP,
        (long long)GRP * QKD, (long long)GRP * QKD, (long long)GRP * GRP);

    // 10) linear-attention KV: kv = (lk^T @ v) / n -> bf16, per batch (128x1024x4096)
    k_gemm<true, false, EPI_BF16><<<dim3(HHALF / 64, QKD / 128, BATCH), B256, 0, stream>>>(
        plk, vb, nullptr, nullptr, kvv, QKD, HHALF, SEQ, QKD, HHALF, HHALF, 1.f / SEQ,
        (long long)SEQ * QKD, (long long)SEQ * HHALF, (long long)QKD * HHALF);
    k_gemm<true, false, EPI_BF16><<<dim3(HHALF / 64, QKD / 128, BATCH), B256, 0, stream>>>(
        plk, ub, nullptr, nullptr, kvu, QKD, HHALF, SEQ, QKD, HHALF, HHALF, 1.f / SEQ,
        (long long)SEQ * QKD, (long long)SEQ * HHALF, (long long)QKD * HHALF);

    // 11) quad part: att = attn @ v (per group, 256x1024x256)
    k_gemm<false, false, EPI_F32><<<dim3(HHALF / 64, GRP / 128, BATCH * GN), B256, 0, stream>>>(
        attn, vb, nullptr, attv, nullptr, GRP, HHALF, GRP, GRP, HHALF, HHALF, 1.f,
        (long long)GRP * GRP, (long long)GRP * HHALF, (long long)GRP * HHALF);
    k_gemm<false, false, EPI_F32><<<dim3(HHALF / 64, GRP / 128, BATCH * GN), B256, 0, stream>>>(
        attn, ub, nullptr, attu, nullptr, GRP, HHALF, GRP, GRP, HHALF, HHALF, 1.f,
        (long long)GRP * GRP, (long long)GRP * HHALF, (long long)GRP * HHALF);

    // 12) linear part accumulated: att += lq @ kv (per batch, 4096x1024x128)
    k_gemm<false, false, EPI_ACC_F32><<<dim3(HHALF / 64, SEQ / 128, BATCH), B256, 0, stream>>>(
        plq, kvv, nullptr, attv, nullptr, SEQ, HHALF, QKD, QKD, HHALF, HHALF, 1.f,
        (long long)SEQ * QKD, (long long)QKD * HHALF, (long long)SEQ * HHALF);
    k_gemm<false, false, EPI_ACC_F32><<<dim3(HHALF / 64, SEQ / 128, BATCH), B256, 0, stream>>>(
        plq, kvu, nullptr, attu, nullptr, SEQ, HHALF, QKD, QKD, HHALF, HHALF, 1.f,
        (long long)SEQ * QKD, (long long)QKD * HHALF, (long long)SEQ * HHALF);

    // 13) gating: outg = att_u * v * sigmoid(att_v * u)
    k_gate<<<dim3((TOK * HHALF) / 256), B256, 0, stream>>>(attv, attu, h2, outg);

    // 14) output LayerNorm (C = 1024) -> bf16
    k_layernorm_bf16<<<dim3(TOK), B256, 0, stream>>>(outg, out_lng, out_lnb, lno, HHALF);

    // 15) out GEMM: silu(lno @ out_W + b)  (16384x512x1024)
    k_gemm<false, false, EPI_BIAS_SILU><<<dim3(DIM / 64, TOK / 128, 1), B256, 0, stream>>>(
        lno, wob, out_b, g3, nullptr, TOK, DIM, HHALF, HHALF, DIM, DIM, 1.f, 0, 0, 0);

    // 16) final ConvModule (LDS-tiled) + residual: y = x + g3 + dwconv(g3)
    k_dwconv_tiled<16><<<dim3(DIM / 256, SEQ / 16, BATCH), B256, 0, stream>>>(
        g3, out_dwk, x, y, nullptr, nullptr, DIM);
}
